// TransformerEncoderLayerTanhRelu_14482629722608
// MI455X (gfx1250) — matrix operand order repaired and checked by reading, not compiled, not run on hardware
//
#include <hip/hip_runtime.h>

// ---------------------------------------------------------------------------
// TransformerEncoderLayer (tanh-relu attention) for MI455X / gfx1250.
// All GEMMs run on v_wmma_f32_16x16x32_bf16 (fp32 accumulate); LN + epilogues
// are VALU. Compute-bound workload (~39 GFLOP vs ~tens of MB of traffic that
// fits in the 192MB L2), so everything funnels to the matrix units.
//
// GEMM: 256-thread block (8 waves) computing a 128x64 tile. The shared W slab
// (64 cols x 32 K, bf16) is staged into a double-buffered LDS tile so the 8
// waves read B-fragments via ds_load_b128; next-step A fragment and W slab
// are fetched into registers while the current WMMAs run (software pipeline),
// one split-barrier per K-step. All four B fragments are materialized before
// the four WMMAs so the ds reads get partial waits instead of full fences.
// Workspace requirement: ~95 MB.
// ---------------------------------------------------------------------------

typedef __attribute__((ext_vector_type(16))) __bf16 v16bf;
typedef __attribute__((ext_vector_type(8)))  __bf16 v8bf;
typedef __attribute__((ext_vector_type(8)))  float  v8f;

constexpr int S  = 2048;
constexpr int B  = 4;
constexpr int D  = 256;
constexpr int H  = 8;
constexpr int HD = 32;
constexpr int F  = 2048;
constexpr int M  = S * B;                         // 8192 token rows
constexpr float QSCALE = 0.17677669529663687f;    // 1/sqrt(HD)

constexpr int LDSPAD = 40;                        // 40 bf16 = 80B row stride

// ---- WMMA fragment loaders (CDNA5 wave32 layouts, CDNA5 ISA §7.12.2) -----
// A (16x32 bf16, MxK): lanes 0-15 = rows, K 0..7 & 16..23; lanes 16-31 same
// rows, K 8..15 & 24..31.  Source row-major with K contiguous.
__device__ __forceinline__ v16bf load_fragA(const __bf16* base, int ld,
                                            int row0, int k0, int lane) {
  const int r    = row0 + (lane & 15);
  const int koff = k0 + ((lane >> 4) << 3);
  const __bf16* p = base + (size_t)r * ld + koff;
  v8bf lo = *(const v8bf*)p;
  v8bf hi = *(const v8bf*)(p + 16);
  v16bf f;
#pragma unroll
  for (int i = 0; i < 8; ++i) { f[i] = lo[i]; f[i + 8] = hi[i]; }
  return f;
}

// B (32x16 bf16, KxN): lane = column; lanes 0-15 hold K 0..7 & 16..23, lanes
// 16-31 hold K 8..15 & 24..31.  Source is "row n holds K contiguous" (W row-major, a
// pre-transposed activation, or an LDS tile in that layout).
__device__ __forceinline__ v16bf load_fragB(const __bf16* base, int ld,
                                            int n0, int k0, int lane) {
  const int n     = n0 + (lane & 15);
  const int kbase = k0 + ((lane >> 4) << 3);
  const __bf16* p = base + (size_t)n * ld + kbase;
  v8bf lo = *(const v8bf*)p;
  v8bf hi = *(const v8bf*)(p + 16);
  v16bf f;
#pragma unroll
  for (int i = 0; i < 8; ++i) { f[i] = lo[i]; f[i + 8] = hi[i]; }
  return f;
}

__device__ __forceinline__ v8f wmma_bf16(v16bf a, v16bf b, v8f c) {
  return __builtin_amdgcn_wmma_f32_16x16x32_bf16(
      /*neg_a=*/false, a, /*neg_b=*/false, b,
      /*c_mod=*/(short)0, c, /*reuse_a=*/false, /*reuse_b=*/false);
}

__device__ __forceinline__ v8f v8f_zero() {
  v8f z = {0.f, 0.f, 0.f, 0.f, 0.f, 0.f, 0.f, 0.f};
  return z;
}

// ---------------------------------------------------------------------------
// Weight fp32 -> bf16 conversion (done once per launch; weights stay in L2).
__global__ __launch_bounds__(256) void cvt_f32_to_bf16(
    const float* __restrict__ src, __bf16* __restrict__ dst, int n) {
  int i = blockIdx.x * blockDim.x + threadIdx.x;
  if (i < n) dst[i] = (__bf16)src[i];
}

// ---------------------------------------------------------------------------
// LayerNorm over D=256: one wave32 per row, 8 elements/lane, shfl reduction.
// Writes fp32 (residual stream) and optionally bf16 (GEMM input).
__global__ __launch_bounds__(256) void ln_kernel(
    const float* __restrict__ x, const float* __restrict__ gamma,
    const float* __restrict__ beta, float* __restrict__ out_f32,
    __bf16* __restrict__ out_bf16) {
  const int lane = threadIdx.x & 31;
  const int row  = blockIdx.x * 8 + (threadIdx.x >> 5);
  const float4* p = (const float4*)(x + (size_t)row * D + lane * 8);
  float4 a = p[0], bq = p[1];
  float v[8] = {a.x, a.y, a.z, a.w, bq.x, bq.y, bq.z, bq.w};
  float s = 0.f, ss = 0.f;
#pragma unroll
  for (int i = 0; i < 8; ++i) { s += v[i]; ss += v[i] * v[i]; }
#pragma unroll
  for (int off = 16; off > 0; off >>= 1) {
    s  += __shfl_xor(s, off, 32);
    ss += __shfl_xor(ss, off, 32);
  }
  const float mean = s * (1.f / D);
  const float var  = ss * (1.f / D) - mean * mean;
  const float rstd = rsqrtf(var + 1e-5f);
#pragma unroll
  for (int i = 0; i < 8; ++i) {
    const int e = lane * 8 + i;
    const float y = (v[i] - mean) * rstd * gamma[e] + beta[e];
    if (out_f32)  out_f32 [(size_t)row * D + e] = y;
    if (out_bf16) out_bf16[(size_t)row * D + e] = (__bf16)y;
  }
}

// ---------------------------------------------------------------------------
// NT GEMM: C[m,n] = act( A[m,:] . W[n,:] + bias[n] ), via WMMA.
// Block = 256 threads = 8 waves; tile 128x64; wave = 16 rows x 4 accumulators.
// mode 0: qkv scatter (q scaled, v transposed);  mode 1: fp32 residual add;
// mode 2: ReLU -> bf16.
__global__ __launch_bounds__(256) void gemm_wmma_kernel(
    const __bf16* __restrict__ A, int lda,
    const __bf16* __restrict__ W, int ldw,
    const float* __restrict__ bias, int N, int K, int mode,
    float* __restrict__ out_f32, __bf16* __restrict__ out_bf16,
    const float* __restrict__ residual,
    __bf16* __restrict__ qd, __bf16* __restrict__ kd, __bf16* __restrict__ vTd) {
  __shared__ __bf16 bs[2][64][LDSPAD];            // 2 x (64 cols x 32 K), padded

  const int tid  = threadIdx.x;
  const int lane = tid & 31;
  const int wave = tid >> 5;
  const int row0 = blockIdx.y * 128 + wave * 16;
  const int col0 = blockIdx.x * 64;

  // Cooperative W-slab staging: thread t covers (col0 + t>>2, k (t&3)*8 ..+7).
  const int sn = tid >> 2;
  const int sk = (tid & 3) * 8;
  const __bf16* wsrc = W + (size_t)(col0 + sn) * ldw + sk;

  v8f acc[4];
#pragma unroll
  for (int j = 0; j < 4; ++j) acc[j] = v8f_zero();

  // Prologue: stage slab 0, load A fragment 0.
  v8bf wreg = *(const v8bf*)wsrc;
  v16bf a_cur = load_fragA(A, lda, row0, 0, lane);
  *(v8bf*)&bs[0][sn][sk] = wreg;
  __syncthreads();

  int buf = 0;
  for (int kk = 0; kk < K; kk += 32) {
    const bool has_next = (kk + 32 < K);
    v16bf a_nxt;
    if (has_next) {
      // Next-step global fetches overlap with this step's WMMAs.
      wreg  = *(const v8bf*)(wsrc + kk + 32);
      a_nxt = load_fragA(A, lda, row0, kk + 32, lane);
      if (kk + 64 < K)  // keep the A stream warm in the near caches
        __builtin_prefetch(A + (size_t)(row0 + (lane & 15)) * lda + kk + 64, 0, 3);
    }

    // Materialize all four B fragments before any WMMA: the ds_load_b128s
    // issue as one batch so the waitcnt pass can use partial dscnt waits and
    // overlap LDS reads with the matrix pipe.
    v16bf b0 = load_fragB(&bs[buf][0][0], LDSPAD, 0,  0, lane);
    v16bf b1 = load_fragB(&bs[buf][0][0], LDSPAD, 16, 0, lane);
    v16bf b2 = load_fragB(&bs[buf][0][0], LDSPAD, 32, 0, lane);
    v16bf b3 = load_fragB(&bs[buf][0][0], LDSPAD, 48, 0, lane);
    acc[0] = wmma_bf16(a_cur, b0, acc[0]);
    acc[1] = wmma_bf16(a_cur, b1, acc[1]);
    acc[2] = wmma_bf16(a_cur, b2, acc[2]);
    acc[3] = wmma_bf16(a_cur, b3, acc[3]);

    if (has_next) {
      *(v8bf*)&bs[buf ^ 1][sn][sk] = wreg;   // other buffer: no hazard w/ reads
      __syncthreads();                       // publish next slab
      buf ^= 1;
      a_cur = a_nxt;
    }
  }

  // C/D layout: vgpr g, lanes 0-15 -> M=g, lanes 16-31 -> M=g+8; N = lane&15.
  const int hi = lane >> 4;
  const int n  = lane & 15;
#pragma unroll
  for (int j = 0; j < 4; ++j) {
    const int c  = col0 + j * 16 + n;
    const float bv = bias[c];
#pragma unroll
    for (int g = 0; g < 8; ++g) {
      const int r = row0 + hi * 8 + g;
      const float val = acc[j][g] + bv;
      if (mode == 0) {
        const int s  = r >> 2, bb = r & 3;          // row = s*B + b, B=4
        const int which = c >> 8, d = c & 255;      // 0=q,1=k,2=v
        const int h = d >> 5, hd = d & 31;
        const int bh = bb * H + h;
        if (which == 0)      qd [((size_t)bh * S + s) * HD + hd] = (__bf16)(val * QSCALE);
        else if (which == 1) kd [((size_t)bh * S + s) * HD + hd] = (__bf16)val;
        else                 vTd[((size_t)bh * HD + hd) * S + s] = (__bf16)val;
      } else if (mode == 1) {
        out_f32[(size_t)r * N + c] = residual[(size_t)r * N + c] + val;
      } else {
        out_bf16[(size_t)r * N + c] = (__bf16)fmaxf(val, 0.f);
      }
    }
  }
}

// ---------------------------------------------------------------------------
// Fused attention: per (b,h) and 16-row q strip, stream over all keys:
//   scores = q.kT (WMMA, K=HD=32), w = relu(tanh(scores)) (VALU),
//   out += w.v (WMMA, K=t in chunks of 32; v pre-transposed to (B,H,HD,S)).
// C-fragment -> A-fragment relayout goes through a per-wave LDS tile.  The
// v fragments are fetched from global BEFORE the activation + LDS round-trip
// (and before the asm memory fence) so their latency hides under the tanh
// VALU work.
__global__ __launch_bounds__(128) void attn_kernel(
    const __bf16* __restrict__ q, const __bf16* __restrict__ kmat,
    const __bf16* __restrict__ vT, __bf16* __restrict__ out) {
  __shared__ __bf16 lds[4][16][LDSPAD];  // 16 rows x 32 probs, padded 80B rows
  const int lane = threadIdx.x & 31;
  const int wave = threadIdx.x >> 5;
  const int s0 = blockIdx.x * 64 + wave * 16;
  const int bh = blockIdx.y;
  const int b  = bh >> 3;
  const int h  = bh & 7;
  const __bf16* qb = q    + (size_t)bh * S * HD;
  const __bf16* kb = kmat + (size_t)bh * S * HD;
  const __bf16* vb = vT   + (size_t)bh * HD * S;

  const v16bf qf = load_fragA(qb, HD, s0, 0, lane);
  const v8f zero = v8f_zero();
  v8f acc0 = v8f_zero(), acc1 = v8f_zero();

  const int hi = lane >> 4;
  const int n  = lane & 15;

  for (int t0 = 0; t0 < S; t0 += 32) {
    v16bf kf0 = load_fragB(kb, HD, t0,      0, lane);
    v16bf kf1 = load_fragB(kb, HD, t0 + 16, 0, lane);
    v8f sc0 = wmma_bf16(qf, kf0, zero);
    v8f sc1 = wmma_bf16(qf, kf1, zero);

    // Independent global loads issued early (above the LDS fence) so they
    // overlap the transcendental VALU work and the LDS relayout.
    v16bf vf0 = load_fragB(vb, S, 0,  t0, lane);
    v16bf vf1 = load_fragB(vb, S, 16, t0, lane);

#pragma unroll
    for (int g = 0; g < 8; ++g) {
      const int m = hi * 8 + g;
      lds[wave][m][n]      = (__bf16)fmaxf(tanhf(sc0[g]), 0.f);
      lds[wave][m][16 + n] = (__bf16)fmaxf(tanhf(sc1[g]), 0.f);
    }
    // per-wave LDS: DS ops are in-order, but keep the compiler honest.
    asm volatile("s_wait_dscnt 0" ::: "memory");
    v16bf prob;
    {
      const int r    = lane & 15;
      const int koff = hi << 3;
#pragma unroll
      for (int i = 0; i < 8; ++i) {
        prob[i]     = lds[wave][r][koff + i];
        prob[i + 8] = lds[wave][r][koff + 16 + i];
      }
    }
    acc0 = wmma_bf16(prob, vf0, acc0);
    acc1 = wmma_bf16(prob, vf1, acc1);
  }

#pragma unroll
  for (int g = 0; g < 8; ++g) {
    const int s = s0 + hi * 8 + g;
    __bf16* o = out + ((size_t)s * B + b) * D + h * HD;
    o[n]      = (__bf16)acc0[g];
    o[16 + n] = (__bf16)acc1[g];
  }
}

// ---------------------------------------------------------------------------
extern "C" void kernel_launch(void* const* d_in, const int* in_sizes, int n_in,
                              void* d_out, int out_size, void* d_ws, size_t ws_size,
                              hipStream_t stream) {
  const float* features = (const float*)d_in[0];
  const float* ln_g     = (const float*)d_in[1];
  const float* ln_b     = (const float*)d_in[2];
  const float* in_w     = (const float*)d_in[3];
  const float* in_b     = (const float*)d_in[4];
  const float* out_w    = (const float*)d_in[5];
  const float* out_b    = (const float*)d_in[6];
  const float* w1       = (const float*)d_in[7];
  const float* b1       = (const float*)d_in[8];
  const float* w2       = (const float*)d_in[9];
  const float* b2       = (const float*)d_in[10];
  float* out = (float*)d_out;

  // Carve workspace (~95 MB total).
  char* ws = (char*)d_ws;
  size_t off = 0;
  auto carve = [&](size_t bytes) -> void* {
    void* p = ws + off;
    off += (bytes + 255) & ~(size_t)255;
    return p;
  };
  float*  lnx_f  = (float*) carve((size_t)M * D * 4);
  __bf16* lnx_h  = (__bf16*)carve((size_t)M * D * 2);
  __bf16* qbuf   = (__bf16*)carve((size_t)B * H * S * HD * 2);
  __bf16* kbuf   = (__bf16*)carve((size_t)B * H * S * HD * 2);
  __bf16* vTbuf  = (__bf16*)carve((size_t)B * H * HD * S * 2);
  __bf16* attn_h = (__bf16*)carve((size_t)M * D * 2);
  float*  x2     = (float*) carve((size_t)M * D * 4);
  float*  r1f    = (float*) carve((size_t)M * D * 4);
  __bf16* r1h    = (__bf16*)carve((size_t)M * D * 2);
  __bf16* h1     = (__bf16*)carve((size_t)M * F * 2);
  float*  x3     = (float*) carve((size_t)M * D * 4);
  __bf16* inw_h  = (__bf16*)carve((size_t)3 * D * D * 2);
  __bf16* outw_h = (__bf16*)carve((size_t)D * D * 2);
  __bf16* w1h    = (__bf16*)carve((size_t)F * D * 2);
  __bf16* w2h    = (__bf16*)carve((size_t)D * F * 2);

  // Weight conversions (fp32 -> bf16).
  cvt_f32_to_bf16<<<(3 * D * D + 255) / 256, 256, 0, stream>>>(in_w,  inw_h,  3 * D * D);
  cvt_f32_to_bf16<<<(D * D + 255) / 256,     256, 0, stream>>>(out_w, outw_h, D * D);
  cvt_f32_to_bf16<<<(F * D + 255) / 256,     256, 0, stream>>>(w1,    w1h,    F * D);
  cvt_f32_to_bf16<<<(D * F + 255) / 256,     256, 0, stream>>>(w2,    w2h,    D * F);

  // ln_x = LN(features)
  ln_kernel<<<M / 8, 256, 0, stream>>>(features, ln_g, ln_b, lnx_f, lnx_h);

  // qkv = ln_x @ in_proj_w.T + in_proj_b  -> scatter q (scaled), k, vT
  gemm_wmma_kernel<<<dim3(3 * D / 64, M / 128), 256, 0, stream>>>(
      lnx_h, D, inw_h, D, in_b, 3 * D, D, /*mode=*/0,
      nullptr, nullptr, nullptr, qbuf, kbuf, vTbuf);

  // attn = relu(tanh(q.kT)) @ v  -> (S,B,D) bf16
  attn_kernel<<<dim3(S / 64, B * H), 128, 0, stream>>>(qbuf, kbuf, vTbuf, attn_h);

  // x2 = ln_x + attn @ out_proj_w.T + out_proj_b
  gemm_wmma_kernel<<<dim3(D / 64, M / 128), 256, 0, stream>>>(
      attn_h, D, outw_h, D, out_b, D, D, /*mode=*/1,
      x2, nullptr, lnx_f, nullptr, nullptr, nullptr);

  // r1 = LN(x2)
  ln_kernel<<<M / 8, 256, 0, stream>>>(x2, ln_g, ln_b, r1f, r1h);

  // h1 = relu(r1 @ w1.T + b1)
  gemm_wmma_kernel<<<dim3(F / 64, M / 128), 256, 0, stream>>>(
      r1h, D, w1h, D, b1, F, D, /*mode=*/2,
      nullptr, h1, nullptr, nullptr, nullptr, nullptr);

  // x3 = r1 + h1 @ w2.T + b2
  gemm_wmma_kernel<<<dim3(D / 64, M / 128), 256, 0, stream>>>(
      h1, F, w2h, F, b2, D, F, /*mode=*/1,
      x3, nullptr, r1f, nullptr, nullptr, nullptr);

  // out = LN(x3)
  ln_kernel<<<M / 8, 256, 0, stream>>>(x3, ln_g, ln_b, out, nullptr);
}
